// sLSTM_68642167325024
// MI455X (gfx1250) — compile-verified, matrix operand order checked by reading, not run on hardware
//
#include <hip/hip_runtime.h>
#include <hip/hip_bf16.h>
#include <math.h>

// Problem constants
#define B_SZ 16384
#define D_SZ 1024
#define H_SZ 1024
#define KCAT 2048   // D + H concatenated reduction dim for GEMM1

typedef __attribute__((ext_vector_type(16))) __bf16 v16bf;
typedef __attribute__((ext_vector_type(8)))  float  v8f;

static __device__ __forceinline__ unsigned short f2bf(float f) {
    unsigned int u = __float_as_uint(f);
    unsigned int r = u + 0x7FFFu + ((u >> 16) & 1u);   // round-to-nearest-even
    return (unsigned short)(r >> 16);
}

// Async copy 16B: global (saddr base + 32-bit byte offset) -> LDS, no VGPR data.
// Tracked by ASYNCcnt; wave out of the way while the async engine moves data.
#define ASYNC_LDS_B128(ldsptr, gbase, byteoff)                                  \
    asm volatile("global_load_async_to_lds_b128 %0, %1, %2"                     \
                 :: "v"((unsigned int)(uintptr_t)(ldsptr)),                     \
                    "v"((unsigned int)(byteoff)),                               \
                    "s"(gbase)                                                  \
                 : "memory")

#define ASYNC_WAIT() asm volatile("s_wait_asynccnt 0x0" ::: "memory")

// ---------------------------------------------------------------------------
// Convert [x | h_prev] -> bf16 A_cat [B, 2048], row-major (K contiguous)
// ---------------------------------------------------------------------------
__global__ __launch_bounds__(256) void k_convert_A(const float* __restrict__ x,
                                                   const float* __restrict__ h,
                                                   unsigned short* __restrict__ A) {
    long t = (long)blockIdx.x * blockDim.x + threadIdx.x;
    long base = t * 4;                       // 4 bf16 per thread
    int row = (int)(base >> 11);
    int k   = (int)(base & 2047);
    const float* src = (k < 1024) ? (x + (long)row * 1024 + k)
                                  : (h + (long)row * 1024 + (k - 1024));
    float4 v = *(const float4*)src;
    unsigned int lo = (unsigned int)f2bf(v.x) | ((unsigned int)f2bf(v.y) << 16);
    unsigned int hi = (unsigned int)f2bf(v.z) | ((unsigned int)f2bf(v.w) << 16);
    *(uint2*)(A + base) = make_uint2(lo, hi);
}

// ---------------------------------------------------------------------------
// Convert gate weights -> bf16 Wt [4096, 2048], N-major (transposed, K contig)
// Wt[n][k]:  n = g*1024 + j ;  k<1024 -> Wg[k][j],  k>=1024 -> Rg[k-1024][j]
// ---------------------------------------------------------------------------
__global__ __launch_bounds__(256) void k_convert_W(
    const float* __restrict__ Wz, const float* __restrict__ Wi,
    const float* __restrict__ Wf, const float* __restrict__ Wo,
    const float* __restrict__ Rz, const float* __restrict__ Ri,
    const float* __restrict__ Rf, const float* __restrict__ Ro,
    unsigned short* __restrict__ Wt) {
    long t = (long)blockIdx.x * blockDim.x + threadIdx.x;
    long base = t * 4;
    int n = (int)(base >> 11);
    int k = (int)(base & 2047);
    int g = n >> 10, j = n & 1023;
    const float* Wp = (g == 0) ? Wz : (g == 1) ? Wi : (g == 2) ? Wf : Wo;
    const float* Rp = (g == 0) ? Rz : (g == 1) ? Ri : (g == 2) ? Rf : Ro;
    const float* src;
    int kk;
    if (k < 1024) { src = Wp; kk = k; } else { src = Rp; kk = k - 1024; }
    unsigned short o[4];
#pragma unroll
    for (int i = 0; i < 4; ++i) o[i] = f2bf(src[(long)(kk + i) * 1024 + j]);
    *(uint2*)(Wt + base) = make_uint2((unsigned int)o[0] | ((unsigned int)o[1] << 16),
                                      (unsigned int)o[2] | ((unsigned int)o[3] << 16));
}

// ---------------------------------------------------------------------------
// Convert Wy -> bf16 WyT [1024, 1024], N-major:  WyT[n][k] = Wy[k][n]
// ---------------------------------------------------------------------------
__global__ __launch_bounds__(256) void k_convert_Wy(const float* __restrict__ Wy,
                                                    unsigned short* __restrict__ WyT) {
    long t = (long)blockIdx.x * blockDim.x + threadIdx.x;
    long base = t * 4;
    int n = (int)(base >> 10);
    int k = (int)(base & 1023);
    unsigned short o[4];
#pragma unroll
    for (int i = 0; i < 4; ++i) o[i] = f2bf(Wy[(long)(k + i) * 1024 + n]);
    *(uint2*)(WyT + base) = make_uint2((unsigned int)o[0] | ((unsigned int)o[1] << 16),
                                       (unsigned int)o[2] | ((unsigned int)o[3] << 16));
}

// ---------------------------------------------------------------------------
// Per-gate-column bias sum: bsum[g*1024 + j] = bW[g][j] + bR[g][j] + b[g][j]
// ---------------------------------------------------------------------------
__global__ __launch_bounds__(256) void k_bias(
    const float* bWz, const float* bRz, const float* bz,
    const float* bWi, const float* bRi, const float* bi,
    const float* bWf, const float* bRf, const float* bfv,
    const float* bWo, const float* bRo, const float* bo,
    float* __restrict__ bsum) {
    int t = blockIdx.x * blockDim.x + threadIdx.x;   // 0..4095
    int g = t >> 10, j = t & 1023;
    const float* a = (g == 0) ? bWz : (g == 1) ? bWi : (g == 2) ? bWf : bWo;
    const float* b = (g == 0) ? bRz : (g == 1) ? bRi : (g == 2) ? bRf : bRo;
    const float* c = (g == 0) ? bz  : (g == 1) ? bi  : (g == 2) ? bfv : bo;
    bsum[t] = a[j] + b[j] + c[j];
}

// ---------------------------------------------------------------------------
// Fused gate GEMM: pre = A_cat @ Wcat  (all 4 gates per workgroup) + sLSTM
// pointwise epilogue.  WG tile: 64 rows x 32 cols x 4 gates.  128 thr = 4 waves.
// Async-to-LDS double-buffered K pipeline.
// ---------------------------------------------------------------------------
__global__ __launch_bounds__(128) void k_gemm_gates(
    const unsigned short* __restrict__ A,    // [B, 2048] bf16
    const unsigned short* __restrict__ Wt,   // [4096, 2048] bf16, N-major
    const float* __restrict__ c_prev,
    const float* __restrict__ n_prev,
    const float* __restrict__ bsum,          // [4096]
    float* __restrict__ out,                 // d_out: y | h | C | n
    unsigned short* __restrict__ hbf)        // [B, 1024] bf16
{
    __shared__ unsigned short lAs[2][64][40];   // 8-short pad -> 80B row stride
    __shared__ unsigned short lBs[2][128][40];

    const int tid  = threadIdx.x;
    const int wave = tid >> 5;
    const int lane = tid & 31;
    const int half = lane >> 4;
    const int l15  = lane & 15;
    const int m0 = blockIdx.x * 64;
    const int j0 = blockIdx.y * 32;

    v8f acc[4][2];
#pragma unroll
    for (int g = 0; g < 4; ++g)
#pragma unroll
        for (int nt = 0; nt < 2; ++nt)
#pragma unroll
            for (int i = 0; i < 8; ++i) acc[g][nt][i] = 0.0f;

    // Per-thread staging chunk coordinates (fixed across K steps)
    int arow[2], aoff[2];
#pragma unroll
    for (int i = 0; i < 2; ++i) {
        int c = tid + i * 128;
        arow[i] = c >> 2; aoff[i] = (c & 3) * 8;
    }
    int bcc[4], boff[4];
    long bnrow[4];
#pragma unroll
    for (int i = 0; i < 4; ++i) {
        int c = tid + i * 128;
        bcc[i] = c >> 2; boff[i] = (c & 3) * 8;
        bnrow[i] = (long)(bcc[i] >> 5) * 1024 + j0 + (bcc[i] & 31);
    }

#define GATES_ISSUE_TILE(k0, buf)                                               \
    do {                                                                        \
        _Pragma("unroll")                                                       \
        for (int i = 0; i < 2; ++i)                                             \
            ASYNC_LDS_B128(&lAs[buf][arow[i]][aoff[i]], A,                      \
                2u * (unsigned)((long)(m0 + arow[i]) * KCAT + (k0) + aoff[i])); \
        _Pragma("unroll")                                                       \
        for (int i = 0; i < 4; ++i)                                             \
            ASYNC_LDS_B128(&lBs[buf][bcc[i]][boff[i]], Wt,                      \
                2u * (unsigned)(bnrow[i] * KCAT + (k0) + boff[i]));             \
    } while (0)

    // Prologue: fill buffer 0
    GATES_ISSUE_TILE(0, 0);
    ASYNC_WAIT();
    __syncthreads();

    const int nsteps = KCAT / 32;
    for (int s = 0; s < nsteps; ++s) {
        const int cur = s & 1, nxt = cur ^ 1;
        if (s + 1 < nsteps) GATES_ISSUE_TILE((s + 1) * 32, nxt);

        // A fragment (16x32 bf16): lane = M, K pairs striped across VGPRs
        union { v16bf v; unsigned int u[8]; } af;
        {
            const unsigned short* base = &lAs[cur][wave * 16 + l15][half * 8];
#pragma unroll
            for (int v = 0; v < 4; ++v) af.u[v]     = *(const unsigned int*)(base + 2 * v);
#pragma unroll
            for (int v = 0; v < 4; ++v) af.u[4 + v] = *(const unsigned int*)(base + 16 + 2 * v);
        }
#pragma unroll
        for (int g = 0; g < 4; ++g) {
#pragma unroll
            for (int nt = 0; nt < 2; ++nt) {
                union { v16bf v; unsigned int u[8]; } bfr;
                const unsigned short* base = &lBs[cur][g * 32 + nt * 16 + l15][half * 8];
#pragma unroll
                for (int v = 0; v < 4; ++v) bfr.u[v]     = *(const unsigned int*)(base + 2 * v);
#pragma unroll
                for (int v = 0; v < 4; ++v) bfr.u[4 + v] = *(const unsigned int*)(base + 16 + 2 * v);
                acc[g][nt] = __builtin_amdgcn_wmma_f32_16x16x32_bf16(
                    false, af.v, false, bfr.v, (short)0, acc[g][nt], false, false);
            }
        }
        ASYNC_WAIT();       // next tile landed in LDS
        __syncthreads();    // everyone done reading cur / writing nxt
    }
#undef GATES_ISSUE_TILE

    // Epilogue: per-lane sLSTM pointwise (all 4 gate accums share (M,N) map)
    const int rbase = m0 + wave * 16 + half * 8;
    float* out_h = out + (long)B_SZ * D_SZ;
    float* out_C = out_h + (long)B_SZ * H_SZ;
    float* out_n = out_C + (long)B_SZ * H_SZ;

#pragma unroll
    for (int nt = 0; nt < 2; ++nt) {
        int col = j0 + nt * 16 + l15;
        float bz_ = bsum[col];
        float bi_ = bsum[1024 + col];
        float bf_ = bsum[2048 + col];
        float bo_ = bsum[3072 + col];
#pragma unroll
        for (int r = 0; r < 8; ++r) {
            int row = rbase + r;
            float pz = acc[0][nt][r] + bz_;
            float pi = acc[1][nt][r] + bi_;
            float pf = acc[2][nt][r] + bf_;
            float po = acc[3][nt][r] + bo_;
            float zt = __builtin_amdgcn_tanhf(pz);           // V_TANH_F32
            float it = __expf(pi);
            float ft = 1.0f / (1.0f + __expf(-pf));
            float ot = 1.0f / (1.0f + __expf(-po));
            long idx = (long)row * H_SZ + col;
            float Cn = ft * c_prev[idx] + it * zt;
            float nn = ft * n_prev[idx] + it;
            float hh = ot * __builtin_amdgcn_tanhf(Cn / (nn + 1e-7f));
            out_h[idx] = hh;
            out_C[idx] = Cn;
            out_n[idx] = nn;
            hbf[idx] = f2bf(hh);
        }
    }
}

// ---------------------------------------------------------------------------
// y = h @ Wy + by    (h in bf16 from ws, WyT N-major bf16)
// WG tile: 64 rows x 32 cols, 128 thr = 4 waves.  Async double-buffered.
// ---------------------------------------------------------------------------
__global__ __launch_bounds__(128) void k_gemm_y(
    const unsigned short* __restrict__ Hb,   // [B, 1024]
    const unsigned short* __restrict__ WyT,  // [1024, 1024]
    const float* __restrict__ by,
    float* __restrict__ y)                   // d_out[0 .. B*D)
{
    __shared__ unsigned short lAs[2][64][40];
    __shared__ unsigned short lBs[2][32][40];

    const int tid  = threadIdx.x;
    const int wave = tid >> 5;
    const int lane = tid & 31;
    const int half = lane >> 4;
    const int l15  = lane & 15;
    const int m0 = blockIdx.x * 64;
    const int j0 = blockIdx.y * 32;

    v8f acc[2];
#pragma unroll
    for (int nt = 0; nt < 2; ++nt)
#pragma unroll
        for (int i = 0; i < 8; ++i) acc[nt][i] = 0.0f;

    int arow[2], aoff[2];
#pragma unroll
    for (int i = 0; i < 2; ++i) {
        int c = tid + i * 128;
        arow[i] = c >> 2; aoff[i] = (c & 3) * 8;
    }
    const int bcc = tid >> 2, boff = (tid & 3) * 8;

#define Y_ISSUE_TILE(k0, buf)                                                   \
    do {                                                                        \
        _Pragma("unroll")                                                       \
        for (int i = 0; i < 2; ++i)                                             \
            ASYNC_LDS_B128(&lAs[buf][arow[i]][aoff[i]], Hb,                     \
                2u * (unsigned)((long)(m0 + arow[i]) * 1024 + (k0) + aoff[i])); \
        ASYNC_LDS_B128(&lBs[buf][bcc][boff], WyT,                               \
            2u * (unsigned)((long)(j0 + bcc) * 1024 + (k0) + boff));            \
    } while (0)

    Y_ISSUE_TILE(0, 0);
    ASYNC_WAIT();
    __syncthreads();

    const int nsteps = 1024 / 32;
    for (int s = 0; s < nsteps; ++s) {
        const int cur = s & 1, nxt = cur ^ 1;
        if (s + 1 < nsteps) Y_ISSUE_TILE((s + 1) * 32, nxt);

        union { v16bf v; unsigned int u[8]; } af;
        {
            const unsigned short* base = &lAs[cur][wave * 16 + l15][half * 8];
#pragma unroll
            for (int v = 0; v < 4; ++v) af.u[v]     = *(const unsigned int*)(base + 2 * v);
#pragma unroll
            for (int v = 0; v < 4; ++v) af.u[4 + v] = *(const unsigned int*)(base + 16 + 2 * v);
        }
#pragma unroll
        for (int nt = 0; nt < 2; ++nt) {
            union { v16bf v; unsigned int u[8]; } bfr;
            const unsigned short* base = &lBs[cur][nt * 16 + l15][half * 8];
#pragma unroll
            for (int v = 0; v < 4; ++v) bfr.u[v]     = *(const unsigned int*)(base + 2 * v);
#pragma unroll
            for (int v = 0; v < 4; ++v) bfr.u[4 + v] = *(const unsigned int*)(base + 16 + 2 * v);
            acc[nt] = __builtin_amdgcn_wmma_f32_16x16x32_bf16(
                false, af.v, false, bfr.v, (short)0, acc[nt], false, false);
        }
        ASYNC_WAIT();
        __syncthreads();
    }
#undef Y_ISSUE_TILE

    const int rbase = m0 + wave * 16 + half * 8;
#pragma unroll
    for (int nt = 0; nt < 2; ++nt) {
        int col = j0 + nt * 16 + l15;
        float bb = by[col];
#pragma unroll
        for (int r = 0; r < 8; ++r) {
            int row = rbase + r;
            y[(long)row * D_SZ + col] = acc[nt][r] + bb;
        }
    }
}

// ---------------------------------------------------------------------------
extern "C" void kernel_launch(void* const* d_in, const int* in_sizes, int n_in,
                              void* d_out, int out_size, void* d_ws, size_t ws_size,
                              hipStream_t stream) {
    const float* x      = (const float*)d_in[0];
    const float* h_prev = (const float*)d_in[1];
    const float* c_prev = (const float*)d_in[2];
    const float* n_prev = (const float*)d_in[3];
    const float* Wz  = (const float*)d_in[4];
    const float* bWz = (const float*)d_in[5];
    const float* Rz  = (const float*)d_in[6];
    const float* bRz = (const float*)d_in[7];
    const float* bz  = (const float*)d_in[8];
    const float* Wi  = (const float*)d_in[9];
    const float* bWi = (const float*)d_in[10];
    const float* Ri  = (const float*)d_in[11];
    const float* bRi = (const float*)d_in[12];
    const float* bi  = (const float*)d_in[13];
    const float* Wf  = (const float*)d_in[14];
    const float* bWf = (const float*)d_in[15];
    const float* Rf  = (const float*)d_in[16];
    const float* bRf = (const float*)d_in[17];
    const float* bfv = (const float*)d_in[18];
    const float* Wo  = (const float*)d_in[19];
    const float* bWo = (const float*)d_in[20];
    const float* Ro  = (const float*)d_in[21];
    const float* bRo = (const float*)d_in[22];
    const float* bo  = (const float*)d_in[23];
    const float* Wy  = (const float*)d_in[24];
    const float* by  = (const float*)d_in[25];

    unsigned char* ws = (unsigned char*)d_ws;
    unsigned short* wsA    = (unsigned short*)(ws);                 //  64 MB
    unsigned short* wsWt   = (unsigned short*)(ws + 67108864);      //  16 MB
    unsigned short* wsWyT  = (unsigned short*)(ws + 83886080);      //   2 MB
    unsigned short* wsH    = (unsigned short*)(ws + 85983232);      //  32 MB
    float*          wsBsum = (float*)(ws + 119537664);              //  16 KB

    // 1) precision conversion / layout transforms
    k_convert_A <<<32768, 256, 0, stream>>>(x, h_prev, wsA);
    k_convert_W <<<8192, 256, 0, stream>>>(Wz, Wi, Wf, Wo, Rz, Ri, Rf, Ro, wsWt);
    k_convert_Wy<<<1024, 256, 0, stream>>>(Wy, wsWyT);
    k_bias      <<<16,   256, 0, stream>>>(bWz, bRz, bz, bWi, bRi, bi,
                                           bWf, bRf, bfv, bWo, bRo, bo, wsBsum);
    // 2) fused 4-gate GEMM + sLSTM pointwise epilogue
    k_gemm_gates<<<dim3(B_SZ / 64, H_SZ / 32), 128, 0, stream>>>(
        wsA, wsWt, c_prev, n_prev, wsBsum, (float*)d_out, wsH);
    // 3) output projection
    k_gemm_y<<<dim3(B_SZ / 64, D_SZ / 32), 128, 0, stream>>>(
        wsH, wsWyT, by, (float*)d_out);
}
